// VariationalGCNEncoder_6743098654921
// MI455X (gfx1250) — matrix operand order repaired and checked by reading, not compiled, not run on hardware
//
#include <hip/hip_runtime.h>
#include <hip/hip_bf16.h>

// Problem constants (match reference)
#define NNODES 50000
#define NEDGES 800000
#define IN_C   10
#define HID    96
#define OUT_C  48
#define MT     4          // m-tiles (of 16 rows) per block in the WMMA GEMMs

typedef __attribute__((ext_vector_type(16))) _Float16 v16h;
typedef __attribute__((ext_vector_type(8)))  float    v8f;

// ---------------------------------------------------------------------------
// Normalization: deg (with self-loop = 1), then dis = rsqrt(deg) in place.
// ---------------------------------------------------------------------------
__global__ void k_deg_init(float* __restrict__ deg, int n) {
    int i = blockIdx.x * blockDim.x + threadIdx.x;
    if (i < n) deg[i] = 1.0f;   // self-loop contribution
}

__global__ void k_deg_edges(const int* __restrict__ col, float* __restrict__ deg, int e) {
    int i = blockIdx.x * blockDim.x + threadIdx.x;
    if (i < e) atomicAdd(&deg[col[i]], 1.0f);
}

__global__ void k_rsqrt_inplace(float* __restrict__ d, int n) {
    int i = blockIdx.x * blockDim.x + threadIdx.x;
    if (i < n) d[i] = rsqrtf(d[i]);   // deg >= 1 always (self-loop)
}

// ---------------------------------------------------------------------------
// Aggregation of x (N x 10): self-loop term is a plain store (runs first),
// edge term is gather + scaled atomic scatter-add.
// ---------------------------------------------------------------------------
__global__ void k_aggx_self(const float* __restrict__ x, const float* __restrict__ dis,
                            float* __restrict__ aggx, int n) {
    int idx = blockIdx.x * blockDim.x + threadIdx.x;
    if (idx < n * IN_C) {
        int i = idx / IN_C;
        float w = dis[i] * dis[i];
        aggx[idx] = x[idx] * w;
    }
}

__global__ void k_aggx_edges(const int* __restrict__ row, const int* __restrict__ col,
                             const float* __restrict__ dis, const float* __restrict__ x,
                             float* __restrict__ aggx, int e) {
    int i = blockIdx.x * blockDim.x + threadIdx.x;
    if (i >= e) return;
    int r = row[i], c = col[i];
    float w = dis[r] * dis[c];
    const float* xs = x + (long)r * IN_C;
    float* o = aggx + (long)c * IN_C;
#pragma unroll
    for (int f = 0; f < IN_C; ++f) atomicAdd(&o[f], xs[f] * w);
}

// ---------------------------------------------------------------------------
// Helper: build the A operand (16x32 f16 tile) for one k-step from a
// row-major [m][K] LDS tile. Contiguous layout: lane<16 -> K 0..15 of
// row M=lane; lane>=16 -> K 16..31.
// ---------------------------------------------------------------------------
__device__ __forceinline__ v16h load_a(const _Float16* As, int rowstride,
                                       int m, int k0, int hi) {
    v16h a;
    const int base = m * rowstride + k0 + 16 * hi;
#pragma unroll
    for (int j = 0; j < 8; ++j) {
        a[j]     = As[base + j];
        a[8 + j] = As[base + 8 + j];
    }
    return a;
}

// ---------------------------------------------------------------------------
// GEMM1: h = relu(aggx @ W1 + b1).  M=N, K=10 (zero-padded to 32), Nout=96.
// Block = 192 threads (6 waves); wave w owns n-tile w (6 tiles cover 96),
// and loops over MT m-tiles of 16 rows. B staged pre-swizzled: per (nt,lane)
// 16 contiguous f16 -> two ds_load_b128 per operand.
// ---------------------------------------------------------------------------
__global__ __launch_bounds__(192) void k_gemm1(const float* __restrict__ aggx,
                                               const float* __restrict__ W1,
                                               const float* __restrict__ b1,
                                               float* __restrict__ h, int nrows) {
    __shared__ _Float16 As[MT * 16 * 32];      // A tiles, row-major [m][k], K padded to 32
    __shared__ _Float16 Bs[6 * 32 * 16];       // swizzled B: [nt][lane][j]
    const int tid = threadIdx.x;
    const int m0  = blockIdx.x * (MT * 16);

    // Stage W1 directly into the per-lane B-operand layout (zero-padded K).
    for (int idx = tid; idx < 6 * 32 * 16; idx += 192) {
        int j    = idx & 15;
        int lane = (idx >> 4) & 31;
        int nt   = idx >> 9;
        int k    = 16 * (lane >> 4) + j;           // K index within the 32-step
        int n    = nt * 16 + (lane & 15);
        Bs[idx] = (k < IN_C) ? (_Float16)W1[k * HID + n] : (_Float16)0.0f;
    }
    for (int idx = tid; idx < MT * 16 * 32; idx += 192) {
        int r = idx >> 5, k = idx & 31;
        int grow = m0 + r; if (grow >= nrows) grow = nrows - 1;   // clamp (keeps EXEC full)
        As[idx] = (k < IN_C) ? (_Float16)aggx[(long)grow * IN_C + k] : (_Float16)0.0f;
    }
    __syncthreads();

    const int wave = tid >> 5, lane = tid & 31;
    const int nb = wave * 16;
    const int hi = lane >> 4, lo = lane & 15;

    const v16h b = *(const v16h*)&Bs[(wave << 9) + (lane << 4)];

    v8f acc[MT];
#pragma unroll
    for (int mt = 0; mt < MT; ++mt) {
        v16h a = load_a(As, 32, mt * 16 + lo, 0, hi);
        v8f c = {};
        acc[mt] = __builtin_amdgcn_wmma_f32_16x16x32_f16(false, a, false, b,
                                                         (short)0, c, false, false);
    }

    const float bias = b1[nb + lo];
#pragma unroll
    for (int mt = 0; mt < MT; ++mt) {
        const int mb = m0 + mt * 16;
        if (mb + 16 <= nrows) {                    // uniform fast path: no per-row guard
#pragma unroll
            for (int r = 0; r < 8; ++r) {          // C/D: VGPR r -> M = r + 8*hi, N = lo
                int m = mb + r + 8 * hi;
                float v = acc[mt][r] + bias;
                h[(long)m * HID + nb + lo] = v > 0.0f ? v : 0.0f;
            }
        } else {
#pragma unroll
            for (int r = 0; r < 8; ++r) {
                int m = mb + r + 8 * hi;
                if (m < nrows) {
                    float v = acc[mt][r] + bias;
                    h[(long)m * HID + nb + lo] = v > 0.0f ? v : 0.0f;
                }
            }
        }
    }
}

// ---------------------------------------------------------------------------
// Aggregation of h (N x 96), shared by mu and logstd layers.
// ---------------------------------------------------------------------------
__global__ void k_aggh_self(const float* __restrict__ h, const float* __restrict__ dis,
                            float* __restrict__ aggh, int n) {
    int idx = blockIdx.x * blockDim.x + threadIdx.x;
    if (idx < n * HID) {
        int i = idx / HID;
        float w = dis[i] * dis[i];
        aggh[idx] = h[idx] * w;
    }
}

#define EPB 16   // edges per block in k_aggh_edges
__global__ __launch_bounds__(HID) void k_aggh_edges(const int* __restrict__ row,
                                                    const int* __restrict__ col,
                                                    const float* __restrict__ dis,
                                                    const float* __restrict__ h,
                                                    float* __restrict__ aggh, int e) {
    const int f = threadIdx.x;           // 96 threads = 3 waves, one feature each
    const int base = blockIdx.x * EPB;
#pragma unroll 1
    for (int k = 0; k < EPB; ++k) {
        int ei = base + k;
        if (ei >= e) return;             // uniform across block
        int r = row[ei], c = col[ei];
        float w = dis[r] * dis[c];
        atomicAdd(&aggh[(long)c * HID + f], h[(long)r * HID + f] * w);
    }
}

// ---------------------------------------------------------------------------
// GEMM2: mu = aggh @ W_mu + b_mu ; logstd = aggh @ W_logstd + b_logstd.
// M=N, K=96 (3 k-steps of 32), Nout=48 (3 n-tiles). Block = 192 threads:
// waves 0..2 -> mu tiles, waves 3..5 -> logstd tiles; A tile shared in LDS.
// k-step outer / m-tile inner: each B operand loaded once feeds MT WMMAs.
// ---------------------------------------------------------------------------
__global__ __launch_bounds__(192) void k_gemm2(const float* __restrict__ aggh,
                                               const float* __restrict__ Wmu,
                                               const float* __restrict__ bmu,
                                               const float* __restrict__ Wls,
                                               const float* __restrict__ bls,
                                               float* __restrict__ out, int nrows) {
    __shared__ _Float16 As[MT * 16 * 96];          // [m][k]
    __shared__ _Float16 Bs[18 * 32 * 16];          // swizzled: [(sel*3+nt)*3+ks][lane][j]
    const int tid = threadIdx.x;
    const int m0  = blockIdx.x * (MT * 16);

    // Stage both W matrices pre-swizzled into the per-lane B-operand layout.
    for (int idx = tid; idx < 18 * 32 * 16; idx += 192) {
        int j    = idx & 15;
        int lane = (idx >> 4) & 31;
        int t    = idx >> 9;               // t = (sel*3 + nt)*3 + ks
        int ks   = t % 3;
        int nt   = (t / 3) % 3;
        int sel  = t / 9;
        int k = ks * 32 + 16 * (lane >> 4) + j;
        int n = nt * 16 + (lane & 15);
        const float* W = sel ? Wls : Wmu;
        Bs[idx] = (_Float16)W[k * OUT_C + n];
    }
    for (int idx = tid; idx < MT * 16 * 96; idx += 192) {
        int r = idx / 96, k = idx % 96;
        int grow = m0 + r; if (grow >= nrows) grow = nrows - 1;
        As[idx] = (_Float16)aggh[(long)grow * HID + k];
    }
    __syncthreads();

    const int wave = tid >> 5, lane = tid & 31;
    const int sel = (wave >= 3) ? 1 : 0;     // 0 = mu, 1 = logstd
    const int nt  = wave % 3;
    const int nb  = nt * 16;
    const int hi  = lane >> 4, lo = lane & 15;

    v8f acc[MT] = {};
#pragma unroll
    for (int ks = 0; ks < 3; ++ks) {
        const v16h b = *(const v16h*)&Bs[((((sel * 3 + nt) * 3) + ks) << 9) + (lane << 4)];
#pragma unroll
        for (int mt = 0; mt < MT; ++mt) {
            v16h a = load_a(As, 96, mt * 16 + lo, ks * 32, hi);
            acc[mt] = __builtin_amdgcn_wmma_f32_16x16x32_f16(false, a, false, b,
                                                             (short)0, acc[mt], false, false);
        }
    }

    const float bias = sel ? bls[nb + lo] : bmu[nb + lo];
    float* dst = out + (sel ? (long)nrows * OUT_C : 0);
#pragma unroll
    for (int mt = 0; mt < MT; ++mt) {
        const int mb = m0 + mt * 16;
        if (mb + 16 <= nrows) {                    // uniform fast path
#pragma unroll
            for (int r = 0; r < 8; ++r) {
                int m = mb + r + 8 * hi;
                dst[(long)m * OUT_C + nb + lo] = acc[mt][r] + bias;
            }
        } else {
#pragma unroll
            for (int r = 0; r < 8; ++r) {
                int m = mb + r + 8 * hi;
                if (m < nrows) dst[(long)m * OUT_C + nb + lo] = acc[mt][r] + bias;
            }
        }
    }
}

// ---------------------------------------------------------------------------
// Host-side launch. Inputs: x, edge_index, W1, b1, W_mu, b_mu, W_logstd, b_logstd
// d_out = [mu (N*48), logstd (N*48)] flat. Workspace layout (floats):
//   dis[N] | aggx[N*10] | h[N*96] | aggh[N*96]   (~40.6 MB)
// ---------------------------------------------------------------------------
extern "C" void kernel_launch(void* const* d_in, const int* in_sizes, int n_in,
                              void* d_out, int out_size, void* d_ws, size_t ws_size,
                              hipStream_t stream) {
    const float* x    = (const float*)d_in[0];
    const int*   ei   = (const int*)d_in[1];      // shape (2, E): row = ei, col = ei + E
    const float* W1   = (const float*)d_in[2];
    const float* b1   = (const float*)d_in[3];
    const float* Wmu  = (const float*)d_in[4];
    const float* bmu  = (const float*)d_in[5];
    const float* Wls  = (const float*)d_in[6];
    const float* bls  = (const float*)d_in[7];
    float* out = (float*)d_out;

    const int* row = ei;
    const int* col = ei + NEDGES;

    float* dis  = (float*)d_ws;
    float* aggx = dis  + NNODES;
    float* h    = aggx + (size_t)NNODES * IN_C;
    float* aggh = h    + (size_t)NNODES * HID;

    const int T = 256;
    const int ROWS_PER_BLK = MT * 16;
    // 1) degree (self-loop init) + edge contributions, then rsqrt in place -> dis
    k_deg_init<<<(NNODES + T - 1) / T, T, 0, stream>>>(dis, NNODES);
    k_deg_edges<<<(NEDGES + T - 1) / T, T, 0, stream>>>(col, dis, NEDGES);
    k_rsqrt_inplace<<<(NNODES + T - 1) / T, T, 0, stream>>>(dis, NNODES);

    // 2) aggregate x, then h = relu(aggx @ W1 + b1) via WMMA
    k_aggx_self<<<(NNODES * IN_C + T - 1) / T, T, 0, stream>>>(x, dis, aggx, NNODES);
    k_aggx_edges<<<(NEDGES + T - 1) / T, T, 0, stream>>>(row, col, dis, x, aggx, NEDGES);
    k_gemm1<<<(NNODES + ROWS_PER_BLK - 1) / ROWS_PER_BLK, 192, 0, stream>>>(aggx, W1, b1, h, NNODES);

    // 3) aggregate h once (shared by mu/logstd), then dual WMMA projection
    k_aggh_self<<<(NNODES * HID + T - 1) / T, T, 0, stream>>>(h, dis, aggh, NNODES);
    k_aggh_edges<<<(NEDGES + EPB - 1) / EPB, HID, 0, stream>>>(row, col, dis, h, aggh, NEDGES);
    k_gemm2<<<(NNODES + ROWS_PER_BLK - 1) / ROWS_PER_BLK, 192, 0, stream>>>(aggh, Wmu, bmu, Wls, bls, out, NNODES);
}
